// DocREModel_29334626632325
// MI455X (gfx1250) — matrix operand order repaired and checked by reading, not compiled, not run on hardware
//
#include <hip/hip_runtime.h>
#include <math.h>

// Problem constants (reference: B,L,H,NH,NE,M = 4,1024,768,12,42,8; EMB=768; OFFSET=1)
#define B_   4
#define L_   1024
#define H_   768
#define NH_  12
#define NE_  42
#define M_   8
#define EMB_ 768

typedef __attribute__((ext_vector_type(16))) __bf16 v16bf;
typedef __attribute__((ext_vector_type(8)))  float  v8f;

__device__ __forceinline__ unsigned short f2bf(float f) {
  unsigned int u = __float_as_uint(f);
  unsigned int r = (u + 0x7FFFu + ((u >> 16) & 1u)) >> 16;  // round-to-nearest-even
  return (unsigned short)r;
}

// ---------------------------------------------------------------------------
// Packed fragment layouts (wave32 WMMA 16x16x32 bf16), element offsets:
//   A[M,K]: off = (tileM*nKt + kt)*512 + lane*16 + i
//     row = lane&15, half = lane>>4; elem i -> k%32 = ((i>>3)<<4) + half*8 + (i&7)
//     inverse: kk=k&31 -> half=(kk>>3)&1, i=((kk>>4)<<3)+(kk&7), lane=half*16+(row&15)
//   B[K,N]: off = (nt*nKt + kt)*512 + lane*16 + i
//     col = lane&15, half = lane>>4; elem i -> k%32 = ((i>>3)<<4) + half*8 + (i&7)
// Each lane's fragment = 16 contiguous bf16 = 32 bytes -> two global b128 loads.
// ---------------------------------------------------------------------------
__device__ __forceinline__ long a_pack_off(int row, int k, int nKt) {
  int kk = k & 31;
  int half = (kk >> 3) & 1;
  int i = ((kk >> 4) << 3) + (kk & 7);
  int lane = half * 16 + (row & 15);
  return ((long)(row >> 4) * nKt + (k >> 5)) * 512 + lane * 16 + i;
}

// ---------------------------------------------------------------------------
// Kernel 1: per-entity logsumexp pooling -> packed-A bf16 [rows=B*NE, K=H]
// ---------------------------------------------------------------------------
__global__ void ent_emb_kernel(const float* __restrict__ seq,
                               const int* __restrict__ mpos,
                               const unsigned char* __restrict__ mmask,
                               unsigned short* __restrict__ entb) {
  int be = blockIdx.x;            // b*NE + e
  int b  = be / NE_;
  const int* pos = mpos + be * M_;
  const unsigned char* mk = mmask + be * M_;
  const int nKt = H_ >> 5;        // 24
  for (int h = threadIdx.x; h < H_; h += blockDim.x) {
    float vals[M_];
    int   vld[M_];
    float mx = -3.4e38f;
    #pragma unroll
    for (int m = 0; m < M_; m++) {
      vld[m] = (mk[m] != 0);
      vals[m] = 0.0f;
      if (vld[m]) {
        int p = pos[m] + 1;       // +OFFSET
        float v = seq[((long)b * L_ + p) * H_ + h];
        vals[m] = v;
        mx = fmaxf(mx, v);
      }
    }
    float s = 0.0f;
    #pragma unroll
    for (int m = 0; m < M_; m++)
      if (vld[m]) s += __expf(vals[m] - mx);
    float o = mx + __logf(s);
    entb[a_pack_off(be, h, nKt)] = f2bf(o);
  }
}

// ---------------------------------------------------------------------------
// Kernel 2: ent_att = mean over valid mentions of attn[b,nh,pos,:]  (fp32)
// ---------------------------------------------------------------------------
__global__ void ent_att_kernel(const float* __restrict__ attn,
                               const int* __restrict__ mpos,
                               const unsigned char* __restrict__ mmask,
                               float* __restrict__ eatt) {
  int blk = blockIdx.x;
  int nh = blk % NH_;
  int e  = (blk / NH_) % NE_;
  int b  = blk / (NH_ * NE_);
  int be = b * NE_ + e;
  const int* pos = mpos + be * M_;
  const unsigned char* mk = mmask + be * M_;
  int l0 = threadIdx.x * 4;
  float4 acc = make_float4(0.f, 0.f, 0.f, 0.f);
  int cnt = 0;
  for (int m = 0; m < M_; m++) {
    if (mk[m]) {
      cnt++;
      int p = pos[m] + 1;
      const float4 v = *(const float4*)(attn + (((long)b * NH_ + nh) * L_ + p) * L_ + l0);
      acc.x += v.x; acc.y += v.y; acc.z += v.z; acc.w += v.w;
    }
  }
  float inv = 1.0f / (float)(cnt > 0 ? cnt : 1);
  acc.x *= inv; acc.y *= inv; acc.z *= inv; acc.w *= inv;
  *(float4*)(eatt + ((long)(be * NH_ + nh)) * L_ + l0) = acc;
}

// ---------------------------------------------------------------------------
// Kernel 3: ht[i,j,:] = relu(sum_h a_i*a_j/NH) normalized over L,
// written as packed-A bf16 (per-batch matrices, rows=NE*NE, K=L).
// ---------------------------------------------------------------------------
#define HT_TILES 112   // ceil(1764/16)=111, padded to 64-row block multiple -> 112
__global__ void ht_kernel(const float* __restrict__ eatt,
                          unsigned short* __restrict__ htn) {
  int blk = blockIdx.x;
  int ej = blk % NE_;
  int ei = (blk / NE_) % NE_;
  int b  = blk / (NE_ * NE_);
  int row = ei * NE_ + ej;
  const float* ai = eatt + ((long)(b * NE_ + ei) * NH_) * L_;
  const float* aj = eatt + ((long)(b * NE_ + ej) * NH_) * L_;
  int l0 = threadIdx.x * 4;
  float4 s = make_float4(0.f, 0.f, 0.f, 0.f);
  #pragma unroll
  for (int h = 0; h < NH_; h++) {
    float4 x = *(const float4*)(ai + h * L_ + l0);
    float4 y = *(const float4*)(aj + h * L_ + l0);
    s.x += x.x * y.x; s.y += x.y * y.y; s.z += x.z * y.z; s.w += x.w * y.w;
  }
  const float invnh = 1.0f / (float)NH_;
  s.x = fmaxf(s.x * invnh, 0.f);
  s.y = fmaxf(s.y * invnh, 0.f);
  s.z = fmaxf(s.z * invnh, 0.f);
  s.w = fmaxf(s.w * invnh, 0.f);

  __shared__ float red[256];
  red[threadIdx.x] = s.x + s.y + s.z + s.w;
  __syncthreads();
  for (int off = 128; off > 0; off >>= 1) {
    if (threadIdx.x < off) red[threadIdx.x] += red[threadIdx.x + off];
    __syncthreads();
  }
  float inv = 1.0f / (red[0] + 1e-10f);

  // pack 4 consecutive k at fixed row: same (kt,half,lane), consecutive i
  const int nKt = L_ >> 5;                       // 32
  int kk   = l0 & 31;
  int half = (kk >> 3) & 1;
  int i0   = ((kk >> 4) << 3) + (kk & 7);        // multiple of 4
  int lane = half * 16 + (row & 15);
  long off = ((long)b * HT_TILES * nKt + (long)(row >> 4) * nKt + (l0 >> 5)) * 512
           + lane * 16 + i0;
  unsigned int w0 = f2bf(s.x * inv) | ((unsigned int)f2bf(s.y * inv) << 16);
  unsigned int w1 = f2bf(s.z * inv) | ((unsigned int)f2bf(s.w * inv) << 16);
  *(uint2*)(htn + off) = make_uint2(w0, w1);
}

// ---------------------------------------------------------------------------
// Kernel 4: pack a row-major f32 [K,N] matrix into packed-B bf16 layout.
// grid.y = batch (strides sSrc/sDst).
// ---------------------------------------------------------------------------
__global__ void pack_b_kernel(const float* __restrict__ src,
                              unsigned short* __restrict__ dst,
                              int K, int N, long sSrc, long sDst) {
  long t4 = ((long)blockIdx.x * blockDim.x + threadIdx.x) * 4;
  long total = (long)K * N;
  if (t4 >= total) return;
  src += (long)blockIdx.y * sSrc;
  dst += (long)blockIdx.y * sDst;
  int nKt = K >> 5;
  int i0   = (int)(t4 & 15);
  int lane = (int)((t4 >> 4) & 31);
  long tt  = t4 >> 9;
  int kt = (int)(tt % nKt);
  int nt = (int)(tt / nKt);
  int c = lane & 15, half = lane >> 4;
  int kb = kt * 32 + ((i0 >> 3) << 4) + half * 8 + (i0 & 7);
  int col = nt * 16 + c;
  unsigned int w0 = f2bf(src[(long)(kb + 0) * N + col])
                  | ((unsigned int)f2bf(src[(long)(kb + 1) * N + col]) << 16);
  unsigned int w1 = f2bf(src[(long)(kb + 2) * N + col])
                  | ((unsigned int)f2bf(src[(long)(kb + 3) * N + col]) << 16);
  *(uint2*)(dst + t4) = make_uint2(w0, w1);
}

// ---------------------------------------------------------------------------
// Packed-operand WMMA GEMM: C = A[M,K] x B[K,N], fp32 accumulate.
// Block = 8 waves arranged 2(M) x 4(N); each wave = 2 Mtiles x 2 Ntiles
// (32x32 per wave, 4 WMMA / K-step, A and B fragments each reused twice).
// Block tile: 64 rows x 128 cols. No LDS, no guarded loads (padded operands).
// Epilogue modes:
//   0: repack fp32 acc as packed-A bf16 into Cb (rows offset by z*rowsPerZ)
//   1: store f32 to Cf
//   2: Cf = tanh(acc + extra[b*NE + i] + bias)  (head -> d_out)
//   3: Cf = tanh(acc + extra[b*NE + j] + bias)  (tail -> d_out)
// ---------------------------------------------------------------------------
__global__ void __launch_bounds__(256)
wmma_gemm_pk(const unsigned short* __restrict__ Ap,
             const unsigned short* __restrict__ Bp,
             int M, int N, int K,
             long sA, long sB, long rowsPerZ,
             const float* __restrict__ bias,
             const float* __restrict__ extra,
             int mode,
             float* __restrict__ Cf,
             unsigned short* __restrict__ Cb) {
  int z = blockIdx.z;
  Ap += (long)z * sA;
  Bp += (long)z * sB;

  int tid  = threadIdx.x;
  int lane = tid & 31;
  int wave = tid >> 5;
  int wm = wave >> 2;                 // 0..1
  int wn = wave & 3;                  // 0..3
  int r    = lane & 15;
  int half = lane >> 4;

  int nKt = K >> 5;
  int tileM0 = blockIdx.y * 4 + wm * 2;      // and +1
  int nt0    = blockIdx.x * 8 + wn * 2;      // and +1

  const unsigned short* ap0 = Ap + (long)tileM0 * nKt * 512 + lane * 16;
  const unsigned short* ap1 = ap0 + (long)nKt * 512;
  const unsigned short* bp0 = Bp + (long)nt0 * nKt * 512 + lane * 16;
  const unsigned short* bp1 = bp0 + (long)nKt * 512;

  v8f acc00 = {};
  v8f acc01 = {};
  v8f acc10 = {};
  v8f acc11 = {};

  #pragma unroll 2
  for (int kt = 0; kt < nKt; kt++) {
    v16bf a0 = *(const v16bf*)ap0;
    v16bf a1 = *(const v16bf*)ap1;
    v16bf b0 = *(const v16bf*)bp0;
    v16bf b1 = *(const v16bf*)bp1;
    acc00 = __builtin_amdgcn_wmma_f32_16x16x32_bf16(false, a0, false, b0,
                                                    (short)0, acc00, false, false);
    acc01 = __builtin_amdgcn_wmma_f32_16x16x32_bf16(false, a0, false, b1,
                                                    (short)0, acc01, false, false);
    acc10 = __builtin_amdgcn_wmma_f32_16x16x32_bf16(false, a1, false, b0,
                                                    (short)0, acc10, false, false);
    acc11 = __builtin_amdgcn_wmma_f32_16x16x32_bf16(false, a1, false, b1,
                                                    (short)0, acc11, false, false);
    ap0 += 512; ap1 += 512; bp0 += 512; bp1 += 512;
  }

  int col0 = nt0 * 16 + r;
  int col1 = col0 + 16;
  int nKtC = N >> 5;                           // for mode-0 repack

  #pragma unroll
  for (int mt = 0; mt < 2; mt++) {
    v8f aL = mt ? acc10 : acc00;
    v8f aR = mt ? acc11 : acc01;
    int rowLoc = (tileM0 + mt) * 16 + half * 8;
    #pragma unroll
    for (int e = 0; e < 8; e++) {
      int row = rowLoc + e;
      if (row >= M) break;
      if (mode == 0) {
        int growp = (int)((long)z * rowsPerZ) + row;  // packed row in global C
        Cb[a_pack_off(growp, col0, nKtC)] = f2bf(aL[e]);
        Cb[a_pack_off(growp, col1, nKtC)] = f2bf(aR[e]);
      } else if (mode == 1) {
        Cf[(long)row * N + col0] = aL[e];
        Cf[(long)row * N + col1] = aR[e];
      } else {
        int bb  = row / (NE_ * NE_);
        int rem = row - bb * NE_ * NE_;
        int sel = (mode == 2) ? (rem / NE_) : (rem % NE_);
        long ex = ((long)(bb * NE_ + sel)) * N;
        Cf[(long)row * N + col0] = tanhf(aL[e] + extra[ex + col0] + bias[col0]);
        Cf[(long)row * N + col1] = tanhf(aR[e] + extra[ex + col1] + bias[col1]);
      }
    }
  }
}

// ---------------------------------------------------------------------------
// Host side
// ---------------------------------------------------------------------------
extern "C" void kernel_launch(void* const* d_in, const int* in_sizes, int n_in,
                              void* d_out, int out_size, void* d_ws, size_t ws_size,
                              hipStream_t stream) {
  (void)in_sizes; (void)n_in; (void)out_size; (void)ws_size;

  const float*         seq   = (const float*)d_in[0];  // [B,L,H]
  const float*         attn  = (const float*)d_in[1];  // [B,NH,L,L]
  const float*         Wh    = (const float*)d_in[2];  // [2H,EMB]
  const float*         bh    = (const float*)d_in[3];  // [EMB]
  const float*         Wt    = (const float*)d_in[4];  // [2H,EMB]
  const float*         bt    = (const float*)d_in[5];  // [EMB]
  const int*           mpos  = (const int*)d_in[6];    // [B,NE,M]
  const unsigned char* mmask = (const unsigned char*)d_in[7]; // [B,NE,M] bool

  char* ws = (char*)d_ws;
  size_t off = 0;
  auto alloc = [&](size_t bytes) -> void* {
    void* p = ws + off;
    off += (bytes + 255) & ~(size_t)255;
    return p;
  };

  const long NEB   = (long)B_ * NE_;          // 168
  const long NPAIR = (long)NE_ * NE_;         // 1764
  const long MPROJ = (long)B_ * NPAIR;        // 7056

  // packed buffer sizes (bf16 elements)
  const long ENTB_E = 12L * 24 * 512;                 // 192 rows pad, K=768
  const long HTN_E  = (long)B_ * HT_TILES * 32 * 512; // per-b 1792 rows pad, K=1024
  const long SEQB_E = (long)B_ * 48 * 32 * 512;       // per-b N=768, K=1024
  const long WHALF_E = 48L * 24 * 512;                // N=768, K=768
  const long RSB_E  = 444L * 24 * 512;                // 7104 rows pad, K=768

  unsigned short* entb = (unsigned short*)alloc(ENTB_E * 2);
  float*          eatt = (float*)         alloc((long)B_ * NE_ * NH_ * L_ * 4);
  unsigned short* htn  = (unsigned short*)alloc(HTN_E * 2);
  unsigned short* seqb = (unsigned short*)alloc(SEQB_E * 2);
  unsigned short* whbT = (unsigned short*)alloc(WHALF_E * 2);
  unsigned short* whbB = (unsigned short*)alloc(WHALF_E * 2);
  unsigned short* wtbT = (unsigned short*)alloc(WHALF_E * 2);
  unsigned short* wtbB = (unsigned short*)alloc(WHALF_E * 2);
  float*          EH   = (float*)         alloc(NEB * EMB_ * 4);
  float*          ET   = (float*)         alloc(NEB * EMB_ * 4);
  unsigned short* rsb  = (unsigned short*)alloc(RSB_E * 2);

  // 1) entity embeddings (logsumexp) -> packed-A bf16
  ent_emb_kernel<<<(int)NEB, 256, 0, stream>>>(seq, mpos, mmask, entb);

  // 2) entity attention (mean over mentions)
  ent_att_kernel<<<B_ * NE_ * NH_, 256, 0, stream>>>(attn, mpos, mmask, eatt);

  // 3) ht gram + relu + normalize -> packed-A bf16 per batch
  ht_kernel<<<B_ * NE_ * NE_, 256, 0, stream>>>(eatt, htn);

  // 4) pack B operands (seq per batch; W head/tail top & bottom halves)
  {
    int n4 = (L_ * H_) / 4;                      // per-batch elements / 4
    pack_b_kernel<<<dim3((n4 + 255) / 256, B_), 256, 0, stream>>>(
        seq, seqb, L_, H_, (long)L_ * H_, 48L * 32 * 512);
    int w4 = (H_ * EMB_) / 4;
    dim3 gw((w4 + 255) / 256, 1);
    pack_b_kernel<<<gw, 256, 0, stream>>>(Wh,                     whbT, H_, EMB_, 0, 0);
    pack_b_kernel<<<gw, 256, 0, stream>>>(Wh + (long)H_ * EMB_,   whbB, H_, EMB_, 0, 0);
    pack_b_kernel<<<gw, 256, 0, stream>>>(Wt,                     wtbT, H_, EMB_, 0, 0);
    pack_b_kernel<<<gw, 256, 0, stream>>>(Wt + (long)H_ * EMB_,   wtbB, H_, EMB_, 0, 0);
  }

  // 5) EH = ent_emb @ W_head_top, ET = ent_emb @ W_tail_top   [168,768]
  {
    dim3 g(EMB_ / 128, 3, 1);   // 3*64 = 192 rows (entb padded to 192)
    wmma_gemm_pk<<<g, 256, 0, stream>>>(entb, whbT, (int)NEB, EMB_, H_,
                                        0, 0, 0, nullptr, nullptr, 1, EH, nullptr);
    wmma_gemm_pk<<<g, 256, 0, stream>>>(entb, wtbT, (int)NEB, EMB_, H_,
                                        0, 0, 0, nullptr, nullptr, 1, ET, nullptr);
  }

  // 6) rs = ht_norm @ seq (batched over B) -> packed-A bf16 rows z*1764+row
  {
    dim3 g(H_ / 128, HT_TILES / 4, B_);   // 28 row-blocks * 64 = 1792 rows pad
    wmma_gemm_pk<<<g, 256, 0, stream>>>(htn, seqb, (int)NPAIR, H_, L_,
                                        (long)HT_TILES * 32 * 512, 48L * 32 * 512,
                                        NPAIR, nullptr, nullptr, 0, nullptr, rsb);
  }

  // 7) h_out = tanh(rs @ W_head_bot + EH[b,i] + b_head)
  //    t_out = tanh(rs @ W_tail_bot + ET[b,j] + b_tail)
  {
    dim3 g(EMB_ / 128, 111, 1);          // 111*64 = 7104 rows (rsb padded)
    float* hout = (float*)d_out;
    float* tout = (float*)d_out + MPROJ * EMB_;
    wmma_gemm_pk<<<g, 256, 0, stream>>>(rsb, whbB, (int)MPROJ, EMB_, H_,
                                        0, 0, 0, bh, EH, 2, hout, nullptr);
    wmma_gemm_pk<<<g, 256, 0, stream>>>(rsb, wtbB, (int)MPROJ, EMB_, H_,
                                        0, 0, 0, bt, ET, 3, tout, nullptr);
  }
}
